// MarlinAttention_17935783428809
// MI455X (gfx1250) — matrix operand order repaired and checked by reading, not compiled, not run on hardware
//
#include <hip/hip_runtime.h>
#include <hip/hip_bf16.h>

typedef _Float16 half_t;
typedef __attribute__((ext_vector_type(16))) _Float16 v16h;
typedef __attribute__((ext_vector_type(8)))  _Float16 v8h;
typedef __attribute__((ext_vector_type(8)))  float    v8f;
typedef int v4i_vs __attribute__((vector_size(16)));   // matches builtin param type

// Model constants
#define HIDDEN 4096
#define NHEADS 32
#define NKVH   8
#define HDIM   128
#define GSIZE  128
#define BATCH  2
#define SEQ    2048
#define MTOK   (BATCH*SEQ)   // 4096 tokens

enum { LAYOUT_PLAIN = 0, LAYOUT_QK = 1, LAYOUT_VT = 2 };

// CDNA5 async global->LDS copy path (ASYNCcnt), if toolchain exposes it.
#if defined(__gfx1250__) && __has_builtin(__builtin_amdgcn_global_load_async_to_lds_b128)
#define HAVE_ASYNC_LDS 1
__device__ __forceinline__ void async_copy_b128(const void* gsrc, void* ldst) {
    __builtin_amdgcn_global_load_async_to_lds_b128(
        (__attribute__((address_space(1))) v4i_vs*)gsrc,
        (__attribute__((address_space(3))) v4i_vs*)ldst, 0, 0);
}
__device__ __forceinline__ void async_wait0() {
#if __has_builtin(__builtin_amdgcn_s_wait_asynccnt)
    __builtin_amdgcn_s_wait_asynccnt(0);
#else
    asm volatile("s_wait_asynccnt 0" ::: "memory");
#endif
}
#else
#define HAVE_ASYNC_LDS 0
#endif

__device__ __forceinline__ v16h cat8(v8h lo, v8h hi) {
    return __builtin_shufflevector(lo, hi, 0,1,2,3,4,5,6,7,8,9,10,11,12,13,14,15);
}

// ---------------------------------------------------------------------------
// Dequant GEMM: out[M x N] = act[M x K] @ ((codes - 8) * scales), K = 4096.
// Workgroup: 256 thr = 8 waves, tile 64(M) x 128(N), K-step 64 (2 WMMA k-substeps).
// Wave (wm = w%4, wn = w/4) computes 16(M) x 64(N) via 4 f32 accumulators.
// LDS A: [m][k] halfs, row stride 72 (144B, 16B-multiple, conflict-free b128)
// LDS B: [n][k] halfs (weight column-major), same stride.
// WMMA A frag (16x32): lane<16: m=lane, k in {0..7,16..23}; lane>=16: {8..15,24..31}
// WMMA B frag (32x16): lane: n=lane%16, k in {0..7,16..23} / {8..15,24..31} by lane/16, same as A
// ---------------------------------------------------------------------------
template<typename AT, typename OT, int LAYOUT>
__global__ __launch_bounds__(256)
void gemm_dq(const AT* __restrict__ act, const int* __restrict__ codes,
             const float* __restrict__ scales, OT* __restrict__ out,
             int N, int H) {
    constexpr int ASTR = 72, BSTR = 72;
    __shared__ half_t Ash[64 * ASTR];
    __shared__ half_t Bsh[128 * BSTR];

    const int tid  = threadIdx.x;
    const int lane = tid & 31;
    const int wave = tid >> 5;
    const int wm   = wave & 3;
    const int wn   = wave >> 2;
    const int m0   = blockIdx.x * 64;
    const int n0   = blockIdx.y * 128;

    const int lrow     = lane & 15;
    const int half_sel = lane >> 4;

    const int bn  = tid & 127;   // weight column handled by this thread
    const int bk0 = tid >> 7;    // 0 or 1

    v8f acc[4] = {};

    for (int k0 = 0; k0 < HIDDEN; k0 += 64) {
        // ---- stage A tile (64 x 64) ----
        if constexpr (__is_same(AT, half_t)) {
            // pure f16 copy: 64 rows x 128B; 2 x b128 per thread
#if HAVE_ASYNC_LDS
            #pragma unroll
            for (int i = 0; i < 2; ++i) {
                const int m   = i * 32 + (tid >> 3);
                const int off = (tid & 7) * 8;           // halfs
                async_copy_b128(act + (size_t)(m0 + m) * HIDDEN + k0 + off,
                                &Ash[m * ASTR + off]);
            }
#else
            #pragma unroll
            for (int i = 0; i < 2; ++i) {
                const int m   = i * 32 + (tid >> 3);
                const int off = (tid & 7) * 8;
                *(v8h*)&Ash[m * ASTR + off] =
                    *(const v8h*)(act + (size_t)(m0 + m) * HIDDEN + k0 + off);
            }
#endif
        } else {
            // f32 -> f16 convert while staging
            #pragma unroll
            for (int i = 0; i < 4; ++i) {
                const int m  = i * 16 + (tid >> 4);
                const int kq = (tid & 15) * 4;
                const AT* src = act + (size_t)(m0 + m) * HIDDEN + k0 + kq;
                half_t* dst = &Ash[m * ASTR + kq];
                #pragma unroll
                for (int j = 0; j < 4; ++j) dst[j] = (half_t)src[j];
            }
        }

        // ---- stage B tile (64 x 128): dequant int4 codes -> f16, store [n][k] ----
        const float sc = scales[(size_t)(k0 >> 7) * N + n0 + bn];
        const int* csrc = codes + (size_t)(k0 + bk0) * N + n0 + bn;
        half_t* bdst = &Bsh[bn * BSTR + bk0];
        // prefetch next iteration's weight block (global_prefetch_b8)
        if (k0 + 64 < HIDDEN) __builtin_prefetch(csrc + (size_t)64 * N, 0, 1);
        #pragma unroll
        for (int r = 0; r < 32; ++r) {
            const int c = csrc[(size_t)(2 * r) * N];
            bdst[2 * r] = (half_t)((float)(c - 8) * sc);
        }
#if HAVE_ASYNC_LDS
        if constexpr (__is_same(AT, half_t)) async_wait0();
#endif
        __syncthreads();

        // ---- fragments + WMMA: 2 k-substeps x 4 n-tiles = 8 WMMAs ----
        #pragma unroll
        for (int ks = 0; ks < 2; ++ks) {
            const half_t* arow = &Ash[(wm * 16 + lrow) * ASTR + ks * 32];
            const v8h a_lo = *(const v8h*)(arow + half_sel * 8);
            const v8h a_hi = *(const v8h*)(arow + 16 + half_sel * 8);
            const v16h afrag = cat8(a_lo, a_hi);
            #pragma unroll
            for (int nt = 0; nt < 4; ++nt) {
                const half_t* brow = &Bsh[(wn * 64 + nt * 16 + lrow) * BSTR + ks * 32];
                const v8h b_lo = *(const v8h*)(brow + half_sel * 8);
                const v8h b_hi = *(const v8h*)(brow + 16 + half_sel * 8);
                const v16h bfrag = cat8(b_lo, b_hi);
                acc[nt] = __builtin_amdgcn_wmma_f32_16x16x32_f16(
                    false, afrag, false, bfrag, (short)0, acc[nt], false, false);
            }
        }
        __syncthreads();
    }

    // ---- epilogue: D layout lane=n%16 col, vgpr r -> row = r + 8*(lane/16) ----
    #pragma unroll
    for (int nt = 0; nt < 4; ++nt) {
        #pragma unroll
        for (int r = 0; r < 8; ++r) {
            const int m = m0 + wm * 16 + r + 8 * half_sel;
            const int n = n0 + wn * 64 + nt * 16 + lrow;
            const float v = acc[nt][r];
            size_t idx;
            if constexpr (LAYOUT == LAYOUT_PLAIN) {
                idx = (size_t)m * N + n;
            } else {
                const int b  = m >> 11;      // / SEQ
                const int s  = m & (SEQ - 1);
                const int h  = n >> 7;       // / HDIM
                const int hd = n & (HDIM - 1);
                if constexpr (LAYOUT == LAYOUT_QK)
                    idx = (((size_t)(b * H + h)) * SEQ + s) * HDIM + hd;
                else // LAYOUT_VT : [b][h][hd][s]
                    idx = (((size_t)(b * H + h)) * HDIM + hd) * SEQ + s;
            }
            out[idx] = (OT)v;
        }
    }
    (void)H;
}

// ---------------------------------------------------------------------------
// RoPE in-place on [B][H][S][HD] f16; one thread per even/odd pair.
// ---------------------------------------------------------------------------
__global__ __launch_bounds__(256)
void rope_kernel(half_t* __restrict__ buf, int total_rows) {
    const int idx = blockIdx.x * blockDim.x + threadIdx.x;
    const int hd2 = idx & 63;
    const int row = idx >> 6;
    if (row >= total_rows) return;
    const int s = row & (SEQ - 1);
    half_t* p = buf + (size_t)row * HDIM + 2 * hd2;
    const float xe = (float)p[0];
    const float xo = (float)p[1];
    // inv_freq = 10000^(-hd2/64) = exp2(-hd2 * log2(10000)/64)
    const float inv = exp2f(-(float)hd2 * (13.287712379549449f / 64.0f));
    const float f = (float)s * inv;
    float sn, cs;
    __sincosf(f, &sn, &cs);
    p[0] = (half_t)(xe * cs - xo * sn);
    p[1] = (half_t)(xo * cs + xe * sn);
}

// ---------------------------------------------------------------------------
// Flash attention: one wave (32 thr) per (b, h, 16-query tile).
// Q: [B][NH][S][HD] f16 (RoPE'd), K: [B][NKV][S][HD] f16 (RoPE'd),
// Vt: [B][NKV][HD][S] f16 (transposed), O: [B*S][NH*HD] f16.
// ---------------------------------------------------------------------------
__global__ __launch_bounds__(32)
void attn_kernel(const half_t* __restrict__ Q, const half_t* __restrict__ Kb,
                 const half_t* __restrict__ Vt, half_t* __restrict__ O) {
    constexpr int QT = SEQ / 16;
    __shared__ half_t Plds[16 * 40];

    const int lane = threadIdx.x & 31;
    const int qt   = blockIdx.x % QT;
    const int h    = (blockIdx.x / QT) % NHEADS;
    const int b    = blockIdx.x / (QT * NHEADS);
    const int kvh  = h / (NHEADS / NKVH);
    const int lrow     = lane & 15;
    const int half_sel = lane >> 4;

    // Load Q fragments (A layout), prescale by 1/sqrt(HD)
    const half_t* Qbase =
        Q + (((size_t)(b * NHEADS + h)) * SEQ + qt * 16 + lrow) * HDIM;
    v16h qfrag[4];
    #pragma unroll
    for (int st = 0; st < 4; ++st) {
        const v8h lo = *(const v8h*)(Qbase + st * 32 + half_sel * 16);
        const v8h hi = *(const v8h*)(Qbase + st * 32 + half_sel * 16 + 8);
        qfrag[st] = cat8(lo, hi) * (half_t)0.08838834764831845f;
    }

    v8f acc[8] = {};
    float mrow[8], lsum[8];
    #pragma unroll
    for (int r = 0; r < 8; ++r) { mrow[r] = -1e30f; lsum[r] = 0.0f; }

    const half_t* Kbase = Kb + ((size_t)(b * NKVH + kvh)) * SEQ * HDIM;
    const half_t* Vbase = Vt + ((size_t)(b * NKVH + kvh)) * HDIM * SEQ;

    const int qmax   = qt * 16 + 15;
    const int nsteps = (qmax + 32) / 32;   // ceil((qmax+1)/32) <= SEQ/32

    for (int kb = 0; kb < nsteps; ++kb) {
        const int key0 = kb * 32;

        // ---- scores: two 16x16 tiles over 32 keys ----
        v8f sc[2] = {};
        #pragma unroll
        for (int t = 0; t < 2; ++t) {
            const half_t* krow = Kbase + (size_t)(key0 + t * 16 + lrow) * HDIM;
            #pragma unroll
            for (int st = 0; st < 4; ++st) {
                // B frag: lane holds column (key) lrow, contiguous 16 d-values
                const v16h kf = *(const v16h*)(krow + st * 32 + half_sel * 16);
                sc[t] = __builtin_amdgcn_wmma_f32_16x16x32_f16(
                    false, qfrag[st], false, kf, (short)0, sc[t], false, false);
            }
        }

        const bool need_mask = (key0 + 31 > qt * 16);
        const float L2E = 1.4426950408889634f;

        // ---- online softmax per row (row = r + 8*half_sel, cols across lanes) ----
        #pragma unroll
        for (int r = 0; r < 8; ++r) {
            const int qg = qt * 16 + r + 8 * half_sel;
            float s0 = sc[0][r];
            float s1 = sc[1][r];
            if (need_mask) {
                if (key0 + lrow > qg)      s0 = -1e30f;
                if (key0 + 16 + lrow > qg) s1 = -1e30f;
            }
            float rmax = fmaxf(s0, s1);
            #pragma unroll
            for (int off = 8; off; off >>= 1)
                rmax = fmaxf(rmax, __shfl_xor(rmax, off, 32));
            const float mnew = fmaxf(mrow[r], rmax);
            const float corr = exp2f((mrow[r] - mnew) * L2E);
            const float p0 = exp2f((s0 - mnew) * L2E);
            const float p1 = exp2f((s1 - mnew) * L2E);
            float rsum = p0 + p1;
            #pragma unroll
            for (int off = 8; off; off >>= 1)
                rsum += __shfl_xor(rsum, off, 32);
            lsum[r] = lsum[r] * corr + rsum;
            mrow[r] = mnew;
            #pragma unroll
            for (int nt = 0; nt < 8; ++nt) acc[nt][r] *= corr;
            const int m = r + 8 * half_sel;
            Plds[m * 40 + lrow]      = (half_t)p0;
            Plds[m * 40 + 16 + lrow] = (half_t)p1;
        }
        __syncthreads();

        // ---- PV: P (16x32) @ Vt tiles (32 x 16 per d-tile) ----
        {
            const half_t* prow = &Plds[lrow * 40];
            const v8h plo = *(const v8h*)(prow + half_sel * 16);
            const v8h phi = *(const v8h*)(prow + half_sel * 16 + 8);
            const v16h pf = cat8(plo, phi);
            #pragma unroll
            for (int nt = 0; nt < 8; ++nt) {
                const half_t* vcol =
                    Vbase + (size_t)(nt * 16 + lrow) * SEQ + key0 + half_sel * 16;
                const v16h vf = *(const v16h*)vcol;
                acc[nt] = __builtin_amdgcn_wmma_f32_16x16x32_f16(
                    false, pf, false, vf, (short)0, acc[nt], false, false);
            }
        }
        __syncthreads();
    }

    // ---- finalize + store [b*S + s][h*HD + d] ----
    #pragma unroll
    for (int r = 0; r < 8; ++r) {
        const float inv = 1.0f / lsum[r];
        const int s = qt * 16 + r + 8 * half_sel;
        half_t* orow = O + ((size_t)(b * SEQ) + s) * (NHEADS * HDIM) + h * HDIM;
        #pragma unroll
        for (int nt = 0; nt < 8; ++nt)
            orow[nt * 16 + lrow] = (half_t)(acc[nt][r] * inv);
    }
}

// ---------------------------------------------------------------------------
extern "C" void kernel_launch(void* const* d_in, const int* in_sizes, int n_in,
                              void* d_out, int out_size, void* d_ws, size_t ws_size,
                              hipStream_t stream) {
    const float* hidden   = (const float*)d_in[0];
    const int*   q_codes  = (const int*)d_in[1];
    const float* q_scales = (const float*)d_in[2];
    const int*   k_codes  = (const int*)d_in[3];
    const float* k_scales = (const float*)d_in[4];
    const int*   v_codes  = (const int*)d_in[5];
    const float* v_scales = (const float*)d_in[6];
    const int*   o_codes  = (const int*)d_in[7];
    const float* o_scales = (const float*)d_in[8];

    const size_t qn = (size_t)BATCH * NHEADS * SEQ * HDIM;  // 16.78M halfs
    const size_t kn = (size_t)BATCH * NKVH  * SEQ * HDIM;   //  4.19M halfs
    half_t* Qb = (half_t*)d_ws;
    half_t* Kbuf = Qb + qn;
    half_t* Vbuf = Kbuf + kn;
    half_t* Ab = Vbuf + kn;                                 // attn out, qn halfs

    const dim3 blk(256);

    // QKV projections (dequant GEMMs, WMMA f16)
    gemm_dq<float, half_t, LAYOUT_QK><<<dim3(MTOK / 64, NHEADS * HDIM / 128), blk, 0, stream>>>(
        hidden, q_codes, q_scales, Qb, NHEADS * HDIM, NHEADS);
    gemm_dq<float, half_t, LAYOUT_QK><<<dim3(MTOK / 64, NKVH * HDIM / 128), blk, 0, stream>>>(
        hidden, k_codes, k_scales, Kbuf, NKVH * HDIM, NKVH);
    gemm_dq<float, half_t, LAYOUT_VT><<<dim3(MTOK / 64, NKVH * HDIM / 128), blk, 0, stream>>>(
        hidden, v_codes, v_scales, Vbuf, NKVH * HDIM, NKVH);

    // RoPE on Q and K
    {
        const int qpairs = (int)(qn / 2);
        const int kpairs = (int)(kn / 2);
        rope_kernel<<<(qpairs + 255) / 256, 256, 0, stream>>>(Qb, BATCH * NHEADS * SEQ);
        rope_kernel<<<(kpairs + 255) / 256, 256, 0, stream>>>(Kbuf, BATCH * NKVH * SEQ);
    }

    // Flash attention: one wave per (b, h, 16-query tile)
    attn_kernel<<<BATCH * NHEADS * (SEQ / 16), 32, 0, stream>>>(Qb, Kbuf, Vbuf, Ab);

    // Output projection -> f32 d_out
    gemm_dq<half_t, float, LAYOUT_PLAIN><<<dim3(MTOK / 64, HIDDEN / 128), blk, 0, stream>>>(
        Ab, o_codes, o_scales, (float*)d_out, HIDDEN, 0);

    (void)in_sizes; (void)n_in; (void)out_size; (void)ws_size;
}
